// SimpleParallelSequentialModel_12799002542576
// MI455X (gfx1250) — matrix operand order repaired and checked by reading, not compiled, not run on hardware
//
#include <hip/hip_runtime.h>

#define DEV __device__ __forceinline__

typedef __bf16 bf16_t;
typedef __attribute__((ext_vector_type(16))) __bf16 v16bf;
typedef __attribute__((ext_vector_type(8)))  __bf16 v8bf;
typedef __attribute__((ext_vector_type(4)))  __bf16 v4bf;
typedef __attribute__((ext_vector_type(8)))  float  v8f;
typedef __attribute__((ext_vector_type(4)))  float  v4f;
typedef __attribute__((ext_vector_type(4)))  unsigned int u32x4;
typedef __attribute__((ext_vector_type(4)))  int i32x4;
typedef __attribute__((ext_vector_type(8)))  int i32x8;

#define HAVE_TDM __has_builtin(__builtin_amdgcn_tensor_load_to_lds)

namespace {
constexpr int kB = 2;
constexpr int kL = 2048;
constexpr int kD = 1024;
constexpr int kV = 32000;
constexpr int kM = kB * kL;   // 4096 token rows
constexpr int kCS = 64;       // scan chunk length
constexpr int kNC = kL / kCS; // 32 chunks per sequence
}

DEV bf16_t f32_to_bf16(float f) {
  union { float f; unsigned u; } in; in.f = f;
  unsigned u = in.u;
  unsigned r = u + 0x7FFFu + ((u >> 16) & 1u);  // round-to-nearest-even
  union { unsigned short s; bf16_t b; } out; out.s = (unsigned short)(r >> 16);
  return out.b;
}

// ---------------------------------------------------------------------------
// f32 -> bf16 conversion, 4 elements/thread
// ---------------------------------------------------------------------------
__global__ void cvt_f32_to_bf16_x4(const float* __restrict__ src,
                                   bf16_t* __restrict__ dst, int n4) {
  int i = blockIdx.x * blockDim.x + threadIdx.x;
  if (i >= n4) return;
  v4f v = *(const v4f*)(src + (size_t)i * 4);
  v4bf o;
  o[0] = f32_to_bf16(v[0]);
  o[1] = f32_to_bf16(v[1]);
  o[2] = f32_to_bf16(v[2]);
  o[3] = f32_to_bf16(v[3]);
  *(v4bf*)(dst + (size_t)i * 4) = o;
}

// ---------------------------------------------------------------------------
// e = emb[input_seq], converted to bf16.  One thread per 4 floats of a row.
// ---------------------------------------------------------------------------
__global__ void gather_embed_bf16(const int* __restrict__ tok,
                                  const float* __restrict__ emb,
                                  bf16_t* __restrict__ out) {
  int i = blockIdx.x * blockDim.x + threadIdx.x;  // over kM * (kD/4)
  int m = i >> 8;            // kD/4 == 256 chunks per row
  int c = (i & 255) << 2;
  int row = tok[m];
  v4f v = *(const v4f*)(emb + (size_t)row * kD + c);
  v4bf o;
  o[0] = f32_to_bf16(v[0]);
  o[1] = f32_to_bf16(v[1]);
  o[2] = f32_to_bf16(v[2]);
  o[3] = f32_to_bf16(v[3]);
  *(v4bf*)(out + (size_t)m * kD + c) = o;
}

// ---------------------------------------------------------------------------
// Causal cumulative mean: 3-phase parallel scan over L (f32), emit bf16.
// ---------------------------------------------------------------------------
__global__ void scan_chunk_sum(const float* __restrict__ Vm,
                               float* __restrict__ sums) {
  int t = blockIdx.x * blockDim.x + threadIdx.x;  // kB*kNC*kD = 65536
  int d = t & (kD - 1);
  int c = (t >> 10) & (kNC - 1);
  int b = t >> 15;
  const float* p = Vm + ((size_t)b * kL + (size_t)c * kCS) * kD + d;
  float s = 0.0f;
#pragma unroll 8
  for (int i = 0; i < kCS; ++i) s += p[(size_t)i * kD];
  sums[t] = s;
}

__global__ void scan_chunk_prefix(float* __restrict__ sums) {
  int t = blockIdx.x * blockDim.x + threadIdx.x;  // kB*kD = 2048
  int d = t & (kD - 1);
  int b = t >> 10;
  float run = 0.0f;
#pragma unroll
  for (int c = 0; c < kNC; ++c) {
    size_t idx = ((size_t)b * kNC + c) * kD + d;
    float s = sums[idx];
    sums[idx] = run;   // exclusive prefix (chunk base)
    run += s;
  }
}

__global__ void scan_finish_bf16(const float* __restrict__ Vm,
                                 const float* __restrict__ sums,
                                 bf16_t* __restrict__ avg) {
  int t = blockIdx.x * blockDim.x + threadIdx.x;  // kB*kNC*kD
  int d = t & (kD - 1);
  int c = (t >> 10) & (kNC - 1);
  int b = t >> 15;
  float acc = sums[t];
  const float* p = Vm + ((size_t)b * kL + (size_t)c * kCS) * kD + d;
  bf16_t* q = avg + ((size_t)b * kL + (size_t)c * kCS) * kD + d;
  const int l0 = c * kCS;
#pragma unroll 8
  for (int i = 0; i < kCS; ++i) {
    acc += p[(size_t)i * kD];
    q[(size_t)i * kD] = f32_to_bf16(acc / (float)(l0 + i + 1));
  }
}

// ---------------------------------------------------------------------------
// TDM helpers: 2D tile (rows x BK bf16) from global into LDS, with hardware
// padding of 16B after every 64B so the LDS row stride is 80B (= BK+8 halves).
// ---------------------------------------------------------------------------
#if HAVE_TDM
DEV i32x8 tdm_group1(int k_elems, int rows) {
  i32x8 g;
  // data_size=1 (2B) | pad_enable | pad_interval=3 (64B) | pad_amount=3 (16B)
  g[0] = (int)((1u << 16) | (1u << 20) | (3u << 22) | (3u << 25));
  g[1] = (int)(((unsigned)k_elems & 0xFFFFu) << 16);          // tensor_dim0 lo
  g[2] = (int)(((unsigned)k_elems >> 16) |
               (((unsigned)rows & 0xFFFFu) << 16));           // dim0 hi | dim1 lo
  g[3] = (int)(((unsigned)rows >> 16) | (32u << 16));         // dim1 hi | tile_dim0=BK
  g[4] = rows;                                                // tile_dim1 (tile_dim2=0)
  g[5] = k_elems;                                             // tensor_dim0_stride lo
  g[6] = 0;
  g[7] = 0;
  return g;
}

DEV void tdm_load_tile(unsigned lds_addr, unsigned long long gaddr, i32x8 g1) {
  u32x4 g0;
  g0[0] = 1u;                                                 // count=1
  g0[1] = lds_addr;
  g0[2] = (unsigned)gaddr;
  g0[3] = (unsigned)((gaddr >> 32) & 0x1FFFFFFull) | (2u << 30);  // type=2
  i32x4 z4 = {0, 0, 0, 0};
#if defined(__clang_major__) && __clang_major__ >= 23
  i32x8 z8 = {0, 0, 0, 0, 0, 0, 0, 0};
  __builtin_amdgcn_tensor_load_to_lds(g0, g1, z4, z4, z8, 0);
#else
  __builtin_amdgcn_tensor_load_to_lds(g0, g1, z4, z4, 0);
#endif
}

DEV unsigned lds_off(const void* p) { return (unsigned)(size_t)p; }
#endif

// ---------------------------------------------------------------------------
// C[M,N] (f32) = A[M,K](bf16) * B[N,K](bf16)^T + bias[N]
// Block: 256 threads (8 waves). Block tile 128x256, BK=32, double-buffered LDS
// fed by the Tensor Data Mover. Wave tile 64x64 = 4x4 wmma sub-tiles.
// ---------------------------------------------------------------------------
union FragBF { v16bf v; v8bf h[2]; };

__global__ __launch_bounds__(256)
void gemm_bf16_nt_wmma(const bf16_t* __restrict__ A,    // [M,K]
                       const bf16_t* __restrict__ Bw,   // [N,K]
                       const float*  __restrict__ bias, // [N] or nullptr
                       float* __restrict__ C,           // [M,N]
                       int M, int N, int K) {
  constexpr int BM = 128, BN = 256, BK = 32;
  constexpr int LDT = BK + 8;  // halves; 80B row stride (16B-aligned rows)
  __shared__ alignas(16) bf16_t As[2][BM][LDT];
  __shared__ alignas(16) bf16_t Bs[2][BN][LDT];

  const int t    = threadIdx.x;
  const int lane = t & 31;
  const int wave = t >> 5;      // 0..7
  const int hlf  = lane >> 4;
  const int lrow = lane & 15;
  const int wm   = wave >> 2;   // 0..1 : 64-row slab
  const int wn   = wave & 3;    // 0..3 : 64-col slab
  const long tileM = (long)blockIdx.y * BM;
  const long tileN = (long)blockIdx.x * BN;

  const bf16_t* Ab = A  + tileM * K;
  const bf16_t* Bb = Bw + tileN * K;

#if HAVE_TDM
  const i32x8 g1A = tdm_group1(K, BM);
  const i32x8 g1B = tdm_group1(K, BN);
  if (wave == 0) {
    tdm_load_tile(lds_off(&As[0][0][0]), (unsigned long long)(size_t)Ab, g1A);
    tdm_load_tile(lds_off(&Bs[0][0][0]), (unsigned long long)(size_t)Bb, g1B);
    __builtin_amdgcn_s_wait_tensorcnt(0);
  }
  __syncthreads();
#else
  // Fallback: stage through VGPRs.  A: 512 16B chunks, B: 1024 16B chunks.
  const int ar0 = t >> 2,           ac0 = (t & 3) << 3;
  const int ar1 = (t + 256) >> 2,   ac1 = ((t + 256) & 3) << 3;
  v8bf ra0, ra1, rb0, rb1, rb2, rb3;
  ra0 = *(const v8bf*)(Ab + (size_t)ar0 * K + ac0);
  ra1 = *(const v8bf*)(Ab + (size_t)ar1 * K + ac1);
  rb0 = *(const v8bf*)(Bb + (size_t)ar0 * K + ac0);
  rb1 = *(const v8bf*)(Bb + (size_t)ar1 * K + ac1);
  rb2 = *(const v8bf*)(Bb + (size_t)(ar0 + 128) * K + ac0);
  rb3 = *(const v8bf*)(Bb + (size_t)(ar1 + 128) * K + ac1);
  *(v8bf*)&As[0][ar0][ac0] = ra0;
  *(v8bf*)&As[0][ar1][ac1] = ra1;
  *(v8bf*)&Bs[0][ar0][ac0] = rb0;
  *(v8bf*)&Bs[0][ar1][ac1] = rb1;
  *(v8bf*)&Bs[0][ar0 + 128][ac0] = rb2;
  *(v8bf*)&Bs[0][ar1 + 128][ac1] = rb3;
  __syncthreads();
#endif

  v8f acc[4][4] = {};

  int cur = 0;
  for (int kt = 0; kt < K; kt += BK) {
    const int nxt = kt + BK;
    const bool more = nxt < K;
#if HAVE_TDM
    if (more && wave == 0) {
      const int nb = cur ^ 1;
      tdm_load_tile(lds_off(&As[nb][0][0]),
                    (unsigned long long)(size_t)(Ab + nxt), g1A);
      tdm_load_tile(lds_off(&Bs[nb][0][0]),
                    (unsigned long long)(size_t)(Bb + nxt), g1B);
    }
#else
    v8bf ta0, ta1, tb0, tb1, tb2, tb3;
    if (more) {
      ta0 = *(const v8bf*)(Ab + (size_t)ar0 * K + nxt + ac0);
      ta1 = *(const v8bf*)(Ab + (size_t)ar1 * K + nxt + ac1);
      tb0 = *(const v8bf*)(Bb + (size_t)ar0 * K + nxt + ac0);
      tb1 = *(const v8bf*)(Bb + (size_t)ar1 * K + nxt + ac1);
      tb2 = *(const v8bf*)(Bb + (size_t)(ar0 + 128) * K + nxt + ac0);
      tb3 = *(const v8bf*)(Bb + (size_t)(ar1 + 128) * K + nxt + ac1);
    }
#endif

    FragBF fa[4], fb[4];
#pragma unroll
    for (int mi = 0; mi < 4; ++mi) {
      int r = wm * 64 + mi * 16 + lrow;
      fa[mi].h[0] = *(const v8bf*)&As[cur][r][hlf * 8];
      fa[mi].h[1] = *(const v8bf*)&As[cur][r][16 + hlf * 8];
    }
#pragma unroll
    for (int ni = 0; ni < 4; ++ni) {
      int r = wn * 64 + ni * 16 + lrow;
      fb[ni].h[0] = *(const v8bf*)&Bs[cur][r][hlf * 8];
      fb[ni].h[1] = *(const v8bf*)&Bs[cur][r][16 + hlf * 8];
    }
#pragma unroll
    for (int mi = 0; mi < 4; ++mi)
#pragma unroll
      for (int ni = 0; ni < 4; ++ni)
        acc[mi][ni] = __builtin_amdgcn_wmma_f32_16x16x32_bf16(
            false, fa[mi].v, false, fb[ni].v, (short)0, acc[mi][ni],
            false, false);

    if (more) {
      const int nb = cur ^ 1;
#if HAVE_TDM
      if (wave == 0) __builtin_amdgcn_s_wait_tensorcnt(0);
#else
      *(v8bf*)&As[nb][ar0][ac0] = ta0;
      *(v8bf*)&As[nb][ar1][ac1] = ta1;
      *(v8bf*)&Bs[nb][ar0][ac0] = tb0;
      *(v8bf*)&Bs[nb][ar1][ac1] = tb1;
      *(v8bf*)&Bs[nb][ar0 + 128][ac0] = tb2;
      *(v8bf*)&Bs[nb][ar1 + 128][ac1] = tb3;
#endif
      __syncthreads();
      cur = nb;
    }
  }

  // Epilogue: bias add + f32 store
#pragma unroll
  for (int ni = 0; ni < 4; ++ni) {
    long col = tileN + wn * 64 + ni * 16 + lrow;
    float bv = bias ? bias[col] : 0.0f;
#pragma unroll
    for (int mi = 0; mi < 4; ++mi) {
      long rowb = tileM + wm * 64 + mi * 16 + hlf * 8;
      float* cp = C + rowb * (long)N + col;
#pragma unroll
      for (int r = 0; r < 8; ++r)
        cp[(long)r * N] = acc[mi][ni][r] + bv;
    }
  }
}

// ---------------------------------------------------------------------------
// Host-side orchestration
// ---------------------------------------------------------------------------
extern "C" void kernel_launch(void* const* d_in, const int* in_sizes, int n_in,
                              void* d_out, int out_size, void* d_ws, size_t ws_size,
                              hipStream_t stream) {
  const int*   tok  = (const int*)d_in[0];    // [2, 2048]
  const float* emb  = (const float*)d_in[1];  // [32000, 1024]
  const float* Wv   = (const float*)d_in[2];  // [1024, 1024]
  const float* Wout = (const float*)d_in[3];  // [32000, 1024]
  const float* bout = (const float*)d_in[4];  // [32000]
  float* out = (float*)d_out;                 // [2, 2048, 32000]

  char* ws = (char*)d_ws;
  size_t o = 0;
  auto carve = [&](size_t bytes) -> void* {
    void* p = ws + o;
    o = (o + bytes + 255) & ~(size_t)255;
    return p;
  };
  bf16_t* e_bf    = (bf16_t*)carve((size_t)kM * kD * sizeof(bf16_t));
  bf16_t* Wv_bf   = (bf16_t*)carve((size_t)kD * kD * sizeof(bf16_t));
  bf16_t* Wout_bf = (bf16_t*)carve((size_t)kV * kD * sizeof(bf16_t));
  float*  Vf      = (float*)carve((size_t)kM * kD * sizeof(float));
  bf16_t* avg_bf  = (bf16_t*)carve((size_t)kM * kD * sizeof(bf16_t));
  float*  sums    = (float*)carve((size_t)kB * kNC * kD * sizeof(float));
  (void)ws_size; (void)in_sizes; (void)n_in; (void)out_size;

  // 1) weight conversions to bf16
  {
    int n4 = kD * kD / 4;
    cvt_f32_to_bf16_x4<<<(n4 + 255) / 256, 256, 0, stream>>>(Wv, Wv_bf, n4);
    n4 = kV * kD / 4;
    cvt_f32_to_bf16_x4<<<(n4 + 255) / 256, 256, 0, stream>>>(Wout, Wout_bf, n4);
  }
  // 2) embedding gather (+ bf16 convert)
  gather_embed_bf16<<<(kM * (kD / 4)) / 256, 256, 0, stream>>>(tok, emb, e_bf);
  // 3) V = e @ W_V^T  (f32 accum/output)
  gemm_bf16_nt_wmma<<<dim3(kD / 256, kM / 128), 256, 0, stream>>>(
      e_bf, Wv_bf, nullptr, Vf, kM, kD, kD);
  // 4) causal cumulative mean -> bf16 (parallel 3-phase scan)
  scan_chunk_sum<<<(kB * kNC * kD) / 256, 256, 0, stream>>>(Vf, sums);
  scan_chunk_prefix<<<(kB * kD) / 256, 256, 0, stream>>>(sums);
  scan_finish_bf16<<<(kB * kNC * kD) / 256, 256, 0, stream>>>(Vf, sums, avg_bf);
  // 5) out = avg @ W_out^T + b_out
  gemm_bf16_nt_wmma<<<dim3(kV / 256, kM / 128), 256, 0, stream>>>(
      avg_bf, Wout_bf, bout, out, kM, kV, kD);
}
